// RBFKanLayer_11879879540919
// MI455X (gfx1250) — matrix operand order repaired and checked by reading, not compiled, not run on hardware
//
#include <hip/hip_runtime.h>
#include <stdint.h>

// ---------------- types for WMMA (gfx1250, wave32) ----------------
typedef __bf16 bf16_t;
typedef bf16_t bf16x2 __attribute__((ext_vector_type(2)));
typedef bf16_t bf16x8 __attribute__((ext_vector_type(8)));   // 16 bytes
typedef bf16_t v16bf  __attribute__((ext_vector_type(16)));  // A/B fragment
typedef float  v8f    __attribute__((ext_vector_type(8)));   // C/D fragment

// ---------------- problem constants ----------------
#define IN_F   1024
#define OUT_F  1024
#define GRID_N 30
#define BATCH  2048
#define K_BASE 1024
#define K_TOT  (K_BASE + IN_F * GRID_N)   // 31744

// ---------------- tiling ----------------
#define BM      64            // rows per workgroup
#define BN      512           // cols per workgroup
#define KC      32            // K per WMMA chunk (bf16 16x16x32)
#define KSPLIT  4
#define KSEG    (K_TOT / KSPLIT)          // 7936, multiple of 32
#define NCH     (KSEG / KC)               // 248 chunks per segment (even)
#define THREADS 512                       // 16 waves of 32
#define A_STRIDE 40                       // KC + 8 pad (bf16 elems): 80B rows, 16B aligned
#define B_STRIDE 40

static_assert((NCH & 1) == 0, "pair-pipelined loop needs an even chunk count");

// f32 -> packed bf16x2 via hardware conversion (v_cvt_pk_bf16_f32)
__device__ __forceinline__ uint32_t pack2(float lo, float hi) {
    bf16x2 p;
    p.x = (bf16_t)lo;
    p.y = (bf16_t)hi;
    union { bf16x2 v; uint32_t u; } c; c.v = p;
    return c.u;
}

// fragment loads (2x ds_load_b128 each) + 8 WMMAs for one K-chunk.
// B fragments ping-pong between two register sets; sched_group_barrier pins
// the interleave (next tile's DS reads issue before the consuming WMMA).
__device__ __forceinline__ void frag_wmma(const uint16_t* aswp, const uint16_t* bswp,
                                          int M, int half, int wn, int l15,
                                          v8f acc[8]) {
    union F { v16bf v; bf16x8 h[2]; };
    F af;
    // A fragment, contiguous k order used here: lanes<16 k0..15 ; lanes>=16 k16..31
    af.h[0] = *reinterpret_cast<const bf16x8*>(&aswp[M * A_STRIDE + half * 16]);
    af.h[1] = *reinterpret_cast<const bf16x8*>(&aswp[M * A_STRIDE + half * 16 + 8]);
    const uint16_t* bbase = &bswp[(wn * 128 + l15) * B_STRIDE + half * 16];
    F b[2];
    // B fragment, same contiguous k order as A: lanes<16 k0..15 ; lanes>=16 k16..31 (n = lane%16)
    b[0].h[0] = *reinterpret_cast<const bf16x8*>(bbase);
    b[0].h[1] = *reinterpret_cast<const bf16x8*>(bbase + 8);
    #pragma unroll
    for (int tt = 0; tt < 8; ++tt) {
        if (tt < 7) {
            b[(tt + 1) & 1].h[0] =
                *reinterpret_cast<const bf16x8*>(bbase + (tt + 1) * 16 * B_STRIDE);
            b[(tt + 1) & 1].h[1] =
                *reinterpret_cast<const bf16x8*>(bbase + (tt + 1) * 16 * B_STRIDE + 8);
        }
        acc[tt] = __builtin_amdgcn_wmma_f32_16x16x32_bf16(
            false, af.v, false, b[tt & 1].v, (short)0, acc[tt], false, false);
    }
    // scheduling pin: 4 DS reads (A + B0), then 7x (WMMA ; 2 DS reads), WMMA
    __builtin_amdgcn_sched_group_barrier(0x100, 4, 0);   // DS_READ x4
    #pragma unroll
    for (int s = 0; s < 7; ++s) {
        __builtin_amdgcn_sched_group_barrier(0x008, 1, 0);   // WMMA
        __builtin_amdgcn_sched_group_barrier(0x100, 2, 0);   // next B frag
    }
    __builtin_amdgcn_sched_group_barrier(0x008, 1, 0);       // last WMMA
}

// stage one K-chunk (A tile + B tile, bf16, fragment-order swizzled) into LDS
__device__ __forceinline__ void stage_chunk(int kb, uint16_t* aswp, uint16_t* bswp,
                                            const float* xrow, const float* bwrow,
                                            const float*& swp, int& ic, int& gc,
                                            int am, int ak0, int t) {
    if (kb < K_BASE) {
        // ---- base region: A = x chunk, B[k][n] = base_weight[n, k] ----
        float4 av = *reinterpret_cast<const float4*>(xrow + kb + ak0);
        uint2 pa; pa.x = pack2(av.x, av.y); pa.y = pack2(av.z, av.w);
        *reinterpret_cast<uint2*>(&aswp[am * A_STRIDE + ak0]) = pa;

        const float4* pb4 = reinterpret_cast<const float4*>(bwrow + kb);
        #pragma unroll
        for (int q = 0; q < 8; ++q) {
            float4 bv = pb4[q];
            uint2 p; p.x = pack2(bv.x, bv.y); p.y = pack2(bv.z, bv.w);
            *reinterpret_cast<uint2*>(&bswp[t * B_STRIDE + q * 4]) = p;
        }
    } else {
        // ---- spline region: A = RBF basis (fused), B = spline_weight ----
        // 4 consecutive ks span at most 2 input features
        float x0 = xrow[ic];
        int   i1 = ic + 1 < IN_F - 1 ? ic + 1 : IN_F - 1;
        float x1 = xrow[i1];
        float vv[4];
        #pragma unroll
        for (int j = 0; j < 4; ++j) {
            int  gj   = gc + j;
            bool wrap = gj >= GRID_N;
            int  g    = wrap ? gj - GRID_N : gj;
            float xv  = wrap ? x1 : x0;
            float d   = xv - (float)g * (1.0f / (float)(GRID_N - 1));
            vv[j] = __expf(-20.0f * d * d);          // v_exp_f32
        }
        // advance chunk state: ks += 32  ->  g += 2 (mod 30), i += 1 (+wrap)
        gc += 2; ic += 1;
        if (gc >= GRID_N) { gc -= GRID_N; ic += 1; }
        uint2 pa; pa.x = pack2(vv[0], vv[1]); pa.y = pack2(vv[2], vv[3]);
        *reinterpret_cast<uint2*>(&aswp[am * A_STRIDE + ak0]) = pa;

        // B[k][n] = spline_weight[k, n]: 32 loads with immediate k*4KB offsets
        #pragma unroll
        for (int q = 0; q < 8; ++q) {
            const int k0 = q * 4;
            float v0 = swp[(k0 + 0) * OUT_F];
            float v1 = swp[(k0 + 1) * OUT_F];
            float v2 = swp[(k0 + 2) * OUT_F];
            float v3 = swp[(k0 + 3) * OUT_F];
            uint2 p; p.x = pack2(v0, v1); p.y = pack2(v2, v3);
            *reinterpret_cast<uint2*>(&bswp[t * B_STRIDE + k0]) = p;
        }
        swp += KC * OUT_F;
    }
}

__global__ void __launch_bounds__(THREADS)
kan_wmma_kernel(const float* __restrict__ x,
                const float* __restrict__ sw,     // [30720, 1024] row-major
                const float* __restrict__ bw,     // [1024, 1024]  (out, in)
                const float* __restrict__ grid,   // [30] == linspace(0,1,30)
                float* __restrict__ out)          // [2048, 1024]
{
    (void)grid;  // grid[g] == g * (1/29), computed arithmetically

    // double-buffered LDS tiles in WMMA fragment order (two b128 per fragment)
    __shared__ __align__(16) uint16_t asw0[BM * A_STRIDE];
    __shared__ __align__(16) uint16_t asw1[BM * A_STRIDE];
    __shared__ __align__(16) uint16_t bsw0[BN * B_STRIDE];
    __shared__ __align__(16) uint16_t bsw1[BN * B_STRIDE];

    const int t    = threadIdx.x;
    const int lane = t & 31;
    const int w    = t >> 5;       // wave 0..15
    const int wm   = w >> 2;       // 0..3  (M tile of this wave)
    const int wn   = w & 3;        // 0..3  (N strip of this wave: 128 cols)

    const int b0     = blockIdx.x * BM;
    const int nb     = blockIdx.y * BN;
    const int kstart = blockIdx.z * KSEG;

    v8f acc[8] = {};   // 8 N-tiles x (16x16 f32) accumulators

    // staging coordinates: A covers 64 rows x 32 k with 512 threads (4 k each)
    const int am  = t >> 3;          // 0..63
    const int ak0 = (t & 7) * 4;     // 0,4,...,28

    const int half = lane >> 4;      // 0: lanes 0-15, 1: lanes 16-31
    const int l15  = lane & 15;
    const int M    = wm * 16 + l15;  // A-fragment row within block

    const float* xrow  = x  + (b0 + am) * IN_F;   // loop-invariant row bases
    const float* bwrow = bw + (nb + t) * IN_F;

    // per-thread incremental (feature i, grid g) for the first spline chunk
    const int kspl0 = kstart < K_BASE ? 0 : kstart - K_BASE;
    int ks0 = kspl0 + ak0;
    int ic  = ks0 / GRID_N;          // one division per kernel
    int gc  = ks0 - ic * GRID_N;
    const float* swp = sw + kspl0 * OUT_F + nb + t;

    // -------- pair-pipelined: compute chunk c while staging chunk c+1 -------
    stage_chunk(kstart, asw0, bsw0, xrow, bwrow, swp, ic, gc, am, ak0, t);

    int kb = kstart;
    for (int c = 0; c < NCH - 2; c += 2) {
        __syncthreads();
        frag_wmma(asw0, bsw0, M, half, wn, l15, acc);
        stage_chunk(kb + KC, asw1, bsw1, xrow, bwrow, swp, ic, gc, am, ak0, t);
        __syncthreads();
        frag_wmma(asw1, bsw1, M, half, wn, l15, acc);
        stage_chunk(kb + 2 * KC, asw0, bsw0, xrow, bwrow, swp, ic, gc, am, ak0, t);
        kb += 2 * KC;
    }
    // final pair, all stage calls unconditional
    __syncthreads();
    frag_wmma(asw0, bsw0, M, half, wn, l15, acc);
    stage_chunk(kb + KC, asw1, bsw1, xrow, bwrow, swp, ic, gc, am, ak0, t);
    __syncthreads();
    frag_wmma(asw1, bsw1, M, half, wn, l15, acc);

    // ---------- epilogue: K-split partials -> float atomics ----------
    const int row0 = b0 + wm * 16;
    #pragma unroll
    for (int tt = 0; tt < 8; ++tt) {
        const int col = nb + wn * 128 + tt * 16 + l15;
        #pragma unroll
        for (int r = 0; r < 8; ++r) {
            const int row = row0 + r + 8 * half;   // C/D VGPR r -> M = r (+8 hi lanes)
            atomicAdd(&out[row * OUT_F + col], acc[tt][r]);
        }
    }
}

__global__ void zero_kernel(float4* __restrict__ p, int n4) {
    int i = blockIdx.x * blockDim.x + threadIdx.x;
    if (i < n4) p[i] = make_float4(0.f, 0.f, 0.f, 0.f);
}

extern "C" void kernel_launch(void* const* d_in, const int* in_sizes, int n_in,
                              void* d_out, int out_size, void* d_ws, size_t ws_size,
                              hipStream_t stream) {
    const float* x    = (const float*)d_in[0];
    const float* sw   = (const float*)d_in[1];
    const float* bw   = (const float*)d_in[2];
    const float* grid = (const float*)d_in[3];
    float* out = (float*)d_out;

    const int n4 = (BATCH * OUT_F) / 4;            // 524288
    zero_kernel<<<n4 / 256, 256, 0, stream>>>((float4*)out, n4);

    dim3 g(BATCH / BM, OUT_F / BN, KSPLIT);        // 32 x 2 x 4 = 256 WGs
    kan_wmma_kernel<<<g, THREADS, 0, stream>>>(x, sw, bw, grid, out);
}
